// CrissCrossAttentionBlock_83829171683584
// MI455X (gfx1250) — matrix operand order repaired and checked by reading, not compiled, not run on hardware
//
#include <hip/hip_runtime.h>
#include <hip/hip_bf16.h>

// ---------------------------------------------------------------------------
// CrissCross attention block for gfx1250 (MI455X), bf16 WMMA pipeline.
// All GEMM operands are arranged so every WMMA fragment load is contiguous
// per lane (16B/32B runs) -> global_load_b128; B fragments of each k-step are
// batch-loaded before the WMMA sequence so the scheduler can drain them with
// incremental s_wait_loadcnt instead of full waits per WMMA.
// ---------------------------------------------------------------------------

typedef __bf16 v16bf __attribute__((ext_vector_type(16)));
typedef float  v8f   __attribute__((ext_vector_type(8)));

constexpr int NB = 8, NH = 96, NW = 96, NC = 512, ND = 64;
constexpr int NS = 96;                       // logits per branch (NH == NW)
constexpr long NPIX = (long)NB * NH * NW;    // 73728
constexpr float NEG_INF_F = -1e30f;

__device__ __forceinline__ v8f zero_v8f() {
  v8f z = {0.f, 0.f, 0.f, 0.f, 0.f, 0.f, 0.f, 0.f};
  return z;
}

__device__ __forceinline__ v8f wmma_bf16(v16bf a, v16bf b, v8f c) {
  return __builtin_amdgcn_wmma_f32_16x16x32_bf16(
      /*neg_a=*/false, a, /*neg_b=*/false, b,
      /*c_mod=*/(short)0, c, /*reuse_a=*/false, /*reuse_b=*/false);
}

// ---- fragment loaders (CDNA5 ISA 7.12.2 VGPR layouts, wave32) --------------
// A 16x32 bf16, row-major source: lanes 0-15 -> M=lane, elems 0..7 = K0..7,
// 8..15 = K16..23; lanes 16-31 -> M=lane-16, K shifted by +8.
__device__ __forceinline__ v16bf load_a_bf(const __bf16* A, long lda,
                                           int m0, int k0, int lane) {
  v16bf a;
  const __bf16* row = A + (long)(m0 + (lane & 15)) * lda + k0 + ((lane >> 4) << 3);
#pragma unroll
  for (int e = 0; e < 16; ++e)
    a[e] = row[(e & 7) + ((e >> 3) << 4)];
  return a;
}

// Same but converting an fp32 source on the fly (x projections).
__device__ __forceinline__ v16bf load_a_f32(const float* A, long lda,
                                            int m0, int k0, int lane) {
  v16bf a;
  const float* row = A + (long)(m0 + (lane & 15)) * lda + k0 + ((lane >> 4) << 3);
#pragma unroll
  for (int e = 0; e < 16; ++e)
    a[e] = (__bf16)row[(e & 7) + ((e >> 3) << 4)];
  return a;
}

// B 32x16 bf16 where the logical B (KxN) is stored transposed as row-major
// N x K (T[n][k]).  lanes 0-15 -> N=lane, elems 0..7 = K0..7, 8..15 = K16..23;
// lanes 16-31 -> N=lane-16, K shifted by +8.  Two 16B runs per lane.
__device__ __forceinline__ v16bf load_bt_bf(const __bf16* T, long ldt,
                                            int k0, int n0, int lane) {
  v16bf b;
  const __bf16* row = T + (long)(n0 + (lane & 15)) * ldt + k0 + ((lane >> 4) << 3);
#pragma unroll
  for (int e = 0; e < 16; ++e) b[e] = row[(e & 7) + ((e >> 3) << 4)];
  return b;
}

// ---------------------------------------------------------------------------
// Kernel 0: fp32 -> bf16 weight conversion WITH transpose: Wt[n][k] = W[k][n].
// ---------------------------------------------------------------------------
__global__ __launch_bounds__(256) void wconv_t_kernel(const float* __restrict__ src,
                                                      __bf16* __restrict__ dst,
                                                      int K, int N) {
  int i = blockIdx.x * 256 + threadIdx.x;
  if (i < K * N) {
    int k = i / N, n = i % N;
    dst[(long)n * K + k] = (__bf16)src[i];
  }
}

// ---------------------------------------------------------------------------
// Kernel 1: fused q+k projection, one pass over x.
// 256 thr = 8 waves; block tile 128 x (64q | 64k). Wave owns one 16-row strip;
// A fragment loaded once per k-step, all 8 B fragments batch-loaded, then 8
// WMMAs drain the loads.
// ---------------------------------------------------------------------------
__global__ __launch_bounds__(256) void gemm_xqk_kernel(const float* __restrict__ X,
                                                       const __bf16* __restrict__ WqT,
                                                       const __bf16* __restrict__ WkT,
                                                       const float* __restrict__ bq,
                                                       const float* __restrict__ bk,
                                                       __bf16* __restrict__ Qb,
                                                       __bf16* __restrict__ Kb) {
  int wave = threadIdx.x >> 5, lane = threadIdx.x & 31;
  int m0 = blockIdx.x * 128 + wave * 16;
  v8f acc[8];
#pragma unroll
  for (int t = 0; t < 8; ++t) acc[t] = zero_v8f();
  for (int k0 = 0; k0 < NC; k0 += 32) {
    v16bf a = load_a_f32(X, NC, m0, k0, lane);
    v16bf bfr[8];
#pragma unroll
    for (int t = 0; t < 4; ++t) bfr[t] = load_bt_bf(WqT, NC, k0, t * 16, lane);
#pragma unroll
    for (int t = 0; t < 4; ++t) bfr[4 + t] = load_bt_bf(WkT, NC, k0, t * 16, lane);
#pragma unroll
    for (int t = 0; t < 8; ++t) acc[t] = wmma_bf16(a, bfr[t], acc[t]);
  }
  int n_in = lane & 15;
  int mb = m0 + ((lane >> 4) << 3);
#pragma unroll
  for (int t = 0; t < 4; ++t) {
    int n = t * 16 + n_in;
    float bbq = bq[n], bbk = bk[n];
#pragma unroll
    for (int r = 0; r < 8; ++r) {
      Qb[(long)(mb + r) * ND + n] = (__bf16)(acc[t][r] + bbq);
      Kb[(long)(mb + r) * ND + n] = (__bf16)(acc[4 + t][r] + bbk);
    }
  }
}

// ---------------------------------------------------------------------------
// Kernel 2: v projection, writing TWO transposed bf16 layouts:
//   vA[b][w][c][h]  (column-attention B operand, contiguous in h)
//   vB[b][h][c][w]  (row-attention B operand,   contiguous in w)
// Scattered u16 stores are absorbed by the 192MB writeback L2 (the whole
// 151MB v footprint stays resident; HBM writeback ~= buffer size).
// NT=16 -> block tile 128x256, grid.y=2 -> x read only twice here.
// B fragments batched 8 at a time to bound register pressure (~200 VGPRs).
// ---------------------------------------------------------------------------
template <int NT>
__global__ __launch_bounds__(256) void gemm_xv_kernel(const float* __restrict__ X,
                                                      const __bf16* __restrict__ WvT,
                                                      const float* __restrict__ bv,
                                                      __bf16* __restrict__ VA,
                                                      __bf16* __restrict__ VB) {
  int wave = threadIdx.x >> 5, lane = threadIdx.x & 31;
  int m0 = blockIdx.x * 128 + wave * 16;
  int n0 = blockIdx.y * (NT * 16);
  v8f acc[NT];
#pragma unroll
  for (int t = 0; t < NT; ++t) acc[t] = zero_v8f();
  for (int k0 = 0; k0 < NC; k0 += 32) {
    v16bf a = load_a_f32(X, NC, m0, k0, lane);
#pragma unroll
    for (int g = 0; g < NT; g += 8) {
      v16bf bfr[8];
#pragma unroll
      for (int t = 0; t < 8; ++t)
        bfr[t] = load_bt_bf(WvT, NC, k0, n0 + (g + t) * 16, lane);
#pragma unroll
      for (int t = 0; t < 8; ++t)
        acc[g + t] = wmma_bf16(a, bfr[t], acc[g + t]);
    }
  }
  int n_in = lane & 15;
  int mb = m0 + ((lane >> 4) << 3);
  // A 16-row tile never crosses an image row (96 % 16 == 0): this lane's 8
  // rows share (b,h) and cover 8 consecutive w.
  int bi = (int)(mb / (NH * NW));
  int rem = (int)(mb % (NH * NW));
  int h = rem / NW, w0 = rem % NW;
  __bf16* vb_base = VB + ((long)(bi * NH + h) * NC) * NW;  // + c*NW + w
  __bf16* va_base = VA + ((long)bi * NW) * NC * NH;        // + (w*NC + c)*NH + h
#pragma unroll
  for (int t = 0; t < NT; ++t) {
    int c = n0 + t * 16 + n_in;
    float bb = bv[c];
    __bf16* vbp = vb_base + (long)c * NW + w0;             // contiguous in r
#pragma unroll
    for (int r = 0; r < 8; ++r) {
      __bf16 val = (__bf16)(acc[t][r] + bb);
      vbp[r] = val;
      va_base[((long)(w0 + r) * NC + c) * NH + h] = val;
    }
  }
}

// ---------------------------------------------------------------------------
// Kernel 3: column scores  SV[b,h,w,g] = sum_c q[b,h,w,c] k[b,g,w,c], diag->-inf
// 192 thr = 6 waves; wave = one 16-row h-tile; the two A fragments are
// preloaded once and reused across all 6 g-tiles. Mask only on diagonal tiles.
// ---------------------------------------------------------------------------
__global__ __launch_bounds__(192) void scores_v_kernel(const __bf16* __restrict__ Q,
                                                       const __bf16* __restrict__ Kb,
                                                       float* __restrict__ SV) {
  int b = blockIdx.x / NW, w = blockIdx.x % NW;
  int wave = threadIdx.x >> 5, lane = threadIdx.x & 31;
  int mi = wave * 16;
  const __bf16* qs = Q + ((long)b * NH * NW + w) * ND;   // row h stride NW*ND
  const __bf16* ks = Kb + ((long)b * NH * NW + w) * ND;
  float* sv = SV + ((long)b * NH * NW + w) * NS;         // elem (h,g): h*NW*NS + g
  const long lda = (long)NW * ND;
  const long ldd = (long)NW * NS;
  v16bf a0 = load_a_bf(qs, lda, mi, 0, lane);
  v16bf a1 = load_a_bf(qs, lda, mi, 32, lane);
  int n_in = lane & 15;
  int mb = mi + ((lane >> 4) << 3);
  for (int nt = 0; nt < 6; ++nt) {
    int ni = nt * 16;
    v16bf b0 = load_bt_bf(ks, lda, 0, ni, lane);
    v16bf b1 = load_bt_bf(ks, lda, 32, ni, lane);
    v8f acc = zero_v8f();
    acc = wmma_bf16(a0, b0, acc);
    acc = wmma_bf16(a1, b1, acc);
    int n = ni + n_in;
    if (mi == ni) {  // wave-uniform: only diagonal tiles need the mask
#pragma unroll
      for (int r = 0; r < 8; ++r) {
        int m = mb + r;
        sv[(long)m * ldd + n] = (m == n) ? NEG_INF_F : acc[r];
      }
    } else {
#pragma unroll
      for (int r = 0; r < 8; ++r)
        sv[(long)(mb + r) * ldd + n] = acc[r];
    }
  }
}

// ---------------------------------------------------------------------------
// Kernel 4: row scores  SH[b,h,w,u] = sum_c q[b,h,w,c] k[b,h,u,c]
// ---------------------------------------------------------------------------
__global__ __launch_bounds__(192) void scores_h_kernel(const __bf16* __restrict__ Q,
                                                       const __bf16* __restrict__ Kb,
                                                       float* __restrict__ SH) {
  int bh = blockIdx.x;  // b*NH + h
  int wave = threadIdx.x >> 5, lane = threadIdx.x & 31;
  int mi = wave * 16;
  const __bf16* qs = Q + (long)bh * NW * ND;
  const __bf16* ks = Kb + (long)bh * NW * ND;
  float* sh = SH + (long)bh * NW * NS;   // elem (w,u): w*NS + u
  v16bf a0 = load_a_bf(qs, ND, mi, 0, lane);
  v16bf a1 = load_a_bf(qs, ND, mi, 32, lane);
  int n_in = lane & 15;
  int mb = mi + ((lane >> 4) << 3);
  for (int nt = 0; nt < 6; ++nt) {
    int ni = nt * 16;
    v16bf b0 = load_bt_bf(ks, ND, 0, ni, lane);
    v16bf b1 = load_bt_bf(ks, ND, 32, ni, lane);
    v8f acc = zero_v8f();
    acc = wmma_bf16(a0, b0, acc);
    acc = wmma_bf16(a1, b1, acc);
    int n = ni + n_in;
#pragma unroll
    for (int r = 0; r < 8; ++r)
      sh[(long)(mb + r) * NS + n] = acc[r];
  }
}

// ---------------------------------------------------------------------------
// Kernel 5: joint softmax over 192 logits/pixel -> bf16 probabilities.
// One wave per pixel, 3 logits/lane/branch, shfl_xor reductions (wave32).
// ---------------------------------------------------------------------------
__global__ __launch_bounds__(256) void softmax_kernel(const float* __restrict__ SV,
                                                      const float* __restrict__ SH,
                                                      __bf16* __restrict__ AV,
                                                      __bf16* __restrict__ AH) {
  long pix = (long)blockIdx.x * 8 + (threadIdx.x >> 5);
  if (pix >= NPIX) return;
  int lane = threadIdx.x & 31;
  const float* svr = SV + pix * NS;
  const float* shr = SH + pix * NS;
  float sv0[3], sh0[3];
  float mx = -3.4e38f;
#pragma unroll
  for (int i = 0; i < 3; ++i) {
    sv0[i] = svr[lane + 32 * i];
    sh0[i] = shr[lane + 32 * i];
    mx = fmaxf(mx, fmaxf(sv0[i], sh0[i]));
  }
#pragma unroll
  for (int off = 16; off > 0; off >>= 1)
    mx = fmaxf(mx, __shfl_xor(mx, off, 32));
  float sum = 0.f;
#pragma unroll
  for (int i = 0; i < 3; ++i) {
    sv0[i] = __expf(sv0[i] - mx);
    sh0[i] = __expf(sh0[i] - mx);
    sum += sv0[i] + sh0[i];
  }
#pragma unroll
  for (int off = 16; off > 0; off >>= 1)
    sum += __shfl_xor(sum, off, 32);
  float inv = 1.f / sum;
#pragma unroll
  for (int i = 0; i < 3; ++i) {
    AV[pix * NS + lane + 32 * i] = (__bf16)(sv0[i] * inv);
    AH[pix * NS + lane + 32 * i] = (__bf16)(sh0[i] * inv);
  }
}

// ---------------------------------------------------------------------------
// Kernel 6: yv GEMM with fused residual: Out = X + gamma * (AV @ V_col)
// B operand from vA[b][w][c][h]: contiguous per-lane fragment loads.
// ---------------------------------------------------------------------------
__global__ __launch_bounds__(192) void out_v_kernel(const __bf16* __restrict__ AV,
                                                    const __bf16* __restrict__ VA,
                                                    const float* __restrict__ X,
                                                    const float* __restrict__ gp,
                                                    float* __restrict__ Out) {
  int b = blockIdx.x / NW, w = blockIdx.x % NW;
  int nc = blockIdx.y * 128;
  int wave = threadIdx.x >> 5, lane = threadIdx.x & 31;
  int m0 = wave * 16;  // h tile
  const __bf16* a = AV + ((long)b * NH * NW + w) * NS;         // row h stride NW*NS
  const __bf16* vt = VA + ((long)b * NW + w) * (long)NC * NH;  // T[c][g], ldt = NH
  const long lda = (long)NW * NS;
  v8f acc[8];
#pragma unroll
  for (int t = 0; t < 8; ++t) acc[t] = zero_v8f();
  for (int k0 = 0; k0 < NS; k0 += 32) {
    v16bf af = load_a_bf(a, lda, m0, k0, lane);
    v16bf bfr[8];
#pragma unroll
    for (int t = 0; t < 8; ++t)
      bfr[t] = load_bt_bf(vt, NH, k0, nc + t * 16, lane);
#pragma unroll
    for (int t = 0; t < 8; ++t)
      acc[t] = wmma_bf16(af, bfr[t], acc[t]);
  }
  float g = gp[0];
  int n_in = lane & 15;
  int mb = m0 + ((lane >> 4) << 3);
#pragma unroll
  for (int t = 0; t < 8; ++t)
#pragma unroll
    for (int r = 0; r < 8; ++r) {
      long idx = ((long)(b * NH + mb + r) * NW + w) * NC + nc + t * 16 + n_in;
      Out[idx] = X[idx] + g * acc[t][r];
    }
}

// ---------------------------------------------------------------------------
// Kernel 7: yh GEMM accumulating into Out: Out += gamma * (AH @ V_row)
// B operand from vB[b][h][c][w]. Stream-ordered after out_v; disjoint lanes.
// ---------------------------------------------------------------------------
__global__ __launch_bounds__(192) void out_h_kernel(const __bf16* __restrict__ AH,
                                                    const __bf16* __restrict__ VB,
                                                    const float* __restrict__ gp,
                                                    float* __restrict__ Out) {
  int bh = blockIdx.x;
  int nc = blockIdx.y * 128;
  int wave = threadIdx.x >> 5, lane = threadIdx.x & 31;
  int m0 = wave * 16;  // w tile
  const __bf16* a = AH + (long)bh * NW * NS;                 // row w stride NS
  const __bf16* vt = VB + (long)bh * (long)NC * NW;          // T[c][u], ldt = NW
  v8f acc[8];
#pragma unroll
  for (int t = 0; t < 8; ++t) acc[t] = zero_v8f();
  for (int k0 = 0; k0 < NS; k0 += 32) {
    v16bf af = load_a_bf(a, NS, m0, k0, lane);
    v16bf bfr[8];
#pragma unroll
    for (int t = 0; t < 8; ++t)
      bfr[t] = load_bt_bf(vt, NW, k0, nc + t * 16, lane);
#pragma unroll
    for (int t = 0; t < 8; ++t)
      acc[t] = wmma_bf16(af, bfr[t], acc[t]);
  }
  float g = gp[0];
  int n_in = lane & 15;
  int mb = m0 + ((lane >> 4) << 3);
#pragma unroll
  for (int t = 0; t < 8; ++t)
#pragma unroll
    for (int r = 0; r < 8; ++r) {
      long idx = ((long)bh * NW + (mb + r)) * NC + nc + t * 16 + n_in;
      Out[idx] += g * acc[t][r];
    }
}

// ---------------------------------------------------------------------------
// Host launcher
// ---------------------------------------------------------------------------
extern "C" void kernel_launch(void* const* d_in, const int* in_sizes, int n_in,
                              void* d_out, int out_size, void* d_ws, size_t ws_size,
                              hipStream_t stream) {
  const float* x     = (const float*)d_in[0];
  const float* Wq    = (const float*)d_in[1];
  const float* bq    = (const float*)d_in[2];
  const float* Wk    = (const float*)d_in[3];
  const float* bk    = (const float*)d_in[4];
  const float* Wv    = (const float*)d_in[5];
  const float* bv    = (const float*)d_in[6];
  const float* gamma = (const float*)d_in[7];
  float* out = (float*)d_out;

  // workspace carve-up (~256 MB total)
  char* ws = (char*)d_ws;
  size_t off = 0;
  auto carve = [&](size_t bytes) {
    void* p = ws + off;
    off += (bytes + 255) & ~(size_t)255;
    return p;
  };
  __bf16* wqT = (__bf16*)carve((size_t)NC * ND * 2);
  __bf16* wkT = (__bf16*)carve((size_t)NC * ND * 2);
  __bf16* wvT = (__bf16*)carve((size_t)NC * NC * 2);
  __bf16* q_b = (__bf16*)carve((size_t)NPIX * ND * 2);
  __bf16* k_b = (__bf16*)carve((size_t)NPIX * ND * 2);
  __bf16* vA  = (__bf16*)carve((size_t)NPIX * NC * 2);
  __bf16* vB  = (__bf16*)carve((size_t)NPIX * NC * 2);
  float*  sv_f = (float*)carve((size_t)NPIX * NS * 4);
  float*  sh_f = (float*)carve((size_t)NPIX * NS * 4);
  __bf16* av_b = (__bf16*)carve((size_t)NPIX * NS * 2);
  __bf16* ah_b = (__bf16*)carve((size_t)NPIX * NS * 2);
  (void)ws_size; (void)in_sizes; (void)n_in; (void)out_size;

  // 0) weights -> transposed bf16
  wconv_t_kernel<<<(NC * ND + 255) / 256, 256, 0, stream>>>(Wq, wqT, NC, ND);
  wconv_t_kernel<<<(NC * ND + 255) / 256, 256, 0, stream>>>(Wk, wkT, NC, ND);
  wconv_t_kernel<<<(NC * NC + 255) / 256, 256, 0, stream>>>(Wv, wvT, NC, NC);

  // 1) projections: fused q+k (one x pass), then v into two transposed layouts
  const int mtiles = (int)(NPIX / 128);  // 576
  gemm_xqk_kernel<<<dim3(mtiles, 1), 256, 0, stream>>>(x, wqT, wkT, bq, bk, q_b, k_b);
  gemm_xv_kernel<16><<<dim3(mtiles, 2), 256, 0, stream>>>(x, wvT, bv, vA, vB);

  // 2) scores (column with fused diagonal mask, then row)
  scores_v_kernel<<<NB * NW, 192, 0, stream>>>(q_b, k_b, sv_f);
  scores_h_kernel<<<NB * NH, 192, 0, stream>>>(q_b, k_b, sh_f);

  // 3) joint softmax over 192 logits per pixel
  softmax_kernel<<<(int)((NPIX + 7) / 8), 256, 0, stream>>>(sv_f, sh_f, av_b, ah_b);

  // 4) outputs: out = x + gamma*yv, then out += gamma*yh
  out_v_kernel<<<dim3(NB * NW, 4), 192, 0, stream>>>(av_b, vA, x, gamma, out);
  out_h_kernel<<<dim3(NB * NH, 4), 192, 0, stream>>>(ah_b, vB, gamma, out);
}
